// SelfAttention2d_9861244912313
// MI455X (gfx1250) — matrix operand order repaired and checked by reading, not compiled, not run on hardware
//
#include <hip/hip_runtime.h>

typedef __attribute__((ext_vector_type(16))) __bf16 v16bf;
typedef __attribute__((ext_vector_type(8)))  float  v8f;

#define HEADS 4
#define DIMH  32
#define NPIX  4096   // 64*64
#define CIN   128
#define INNER 128
#define BATCH 4
#define SCALE 0.17677669529663687f  // 32^-0.5

// ---------- WMMA helper ----------
__device__ __forceinline__ v8f wmma_bf16(v16bf a, v16bf b, v8f c) {
  return __builtin_amdgcn_wmma_f32_16x16x32_bf16(false, a, false, b, (short)0, c,
                                                 false, false);
}

// B-operand loader: 16 bf16 = 32 bytes contiguous (k = 16h .. 16h+15).
__device__ __forceinline__ v16bf ld_bf16x16(const __bf16* p) {
  union { v16bf v; uint4 q[2]; } u;
  const uint4* s = (const uint4*)p;
  u.q[0] = s[0];
  u.q[1] = s[1];
  return u.v;
}

// A-operand loader: two separate 16-byte chunks (k = {16h..16h+7}, {16h+8..16h+15}).
__device__ __forceinline__ v16bf ld_bf16_2x8(const __bf16* p0, const __bf16* p1) {
  union { v16bf v; uint4 q[2]; } u;
  u.q[0] = *(const uint4*)p0;
  u.q[1] = *(const uint4*)p1;
  return u.v;
}

// A-operand from fp32 weights: two 8-float chunks, converted to bf16.
__device__ __forceinline__ v16bf ld_f32_to_bf16(const float* p0, const float* p1) {
  union { v16bf v; __bf16 h[16]; } u;
  const float4* a = (const float4*)p0;
  const float4* b = (const float4*)p1;
  float4 a0 = a[0], a1 = a[1], b0 = b[0], b1 = b[1];
  u.h[0] = (__bf16)a0.x;  u.h[1] = (__bf16)a0.y;  u.h[2] = (__bf16)a0.z;  u.h[3] = (__bf16)a0.w;
  u.h[4] = (__bf16)a1.x;  u.h[5] = (__bf16)a1.y;  u.h[6] = (__bf16)a1.z;  u.h[7] = (__bf16)a1.w;
  u.h[8] = (__bf16)b0.x;  u.h[9] = (__bf16)b0.y;  u.h[10] = (__bf16)b0.z; u.h[11] = (__bf16)b0.w;
  u.h[12] = (__bf16)b1.x; u.h[13] = (__bf16)b1.y; u.h[14] = (__bf16)b1.z; u.h[15] = (__bf16)b1.w;
  return u.v;
}

// Compiler-only reordering fence. The P LDS region is wave-private and LDS
// instructions of one wave execute in order (DScnt, in-order per ISA), so no
// s_barrier is needed for the store->cross-lane-load exchange.
__device__ __forceinline__ void lds_compiler_fence() {
  asm volatile("" ::: "memory");
}

// ================= Phase 1: QKV projection (1x1 conv as GEMM) =================
// grid (64 p-blocks, 3 sections, B), block 256 (8 waves).
// Section 0 -> Q (scaled, row-major [bh,n,d]); 1 -> K (row-major); 2 -> V (transposed [bh,d,n]).
__global__ void __launch_bounds__(256)
qkv_kernel(const float* __restrict__ x, const float* __restrict__ wqkv,
           __bf16* __restrict__ Qs, __bf16* __restrict__ Ks, __bf16* __restrict__ Vt) {
  __shared__ __bf16 Xt[64 * 32];  // x tile, transposed to [p][c], bf16
  const int t    = threadIdx.x;
  const int wav  = t >> 5;
  const int lane = t & 31;
  const int col  = lane & 15;
  const int half = lane >> 4;
  const int p0   = blockIdx.x * 64;
  const int sec  = blockIdx.y;
  const int b    = blockIdx.z;

  v8f acc[4] = {};

  for (int kc = 0; kc < CIN; kc += 32) {
    __syncthreads();
    {  // cooperative load+transpose: x[b, kc+cl, p0+8pc .. +8] -> Xt[p][c]
      const int cl = t & 31;
      const int pc = t >> 5;
      const float4* src =
          (const float4*)(x + ((size_t)b * CIN + kc + cl) * NPIX + p0 + pc * 8);
      float4 f0 = src[0], f1 = src[1];
      __bf16* dst = Xt + (pc * 8) * 32 + cl;
      dst[0 * 32] = (__bf16)f0.x; dst[1 * 32] = (__bf16)f0.y;
      dst[2 * 32] = (__bf16)f0.z; dst[3 * 32] = (__bf16)f0.w;
      dst[4 * 32] = (__bf16)f1.x; dst[5 * 32] = (__bf16)f1.y;
      dst[6 * 32] = (__bf16)f1.z; dst[7 * 32] = (__bf16)f1.w;
    }
    __syncthreads();

    const float* wrow = wqkv + (size_t)(sec * 128 + wav * 16 + col) * CIN + kc;
    v16bf A = ld_f32_to_bf16(wrow + 16 * half, wrow + 16 * half + 8);
#pragma unroll
    for (int j = 0; j < 4; ++j) {
      v16bf B = ld_bf16x16(Xt + (16 * j + col) * 32 + 16 * half);
      acc[j] = wmma_bf16(A, B, acc[j]);
    }
  }

#pragma unroll
  for (int j = 0; j < 4; ++j) {
#pragma unroll
    for (int g = 0; g < 8; ++g) {
      const int ol = wav * 16 + g + 8 * half;  // output channel within section
      const int hh = ol >> 5;                  // head
      const int d  = ol & 31;                  // dim within head
      const int p  = p0 + 16 * j + col;
      const float v = acc[j][g];
      const size_t bh = (size_t)(b * HEADS + hh);
      if (sec == 0)
        Qs[(bh * NPIX + p) * DIMH + d] = (__bf16)(v * SCALE);
      else if (sec == 1)
        Ks[(bh * NPIX + p) * DIMH + d] = (__bf16)v;
      else
        Vt[(bh * DIMH + d) * NPIX + p] = (__bf16)v;
    }
  }
}

// ================= Phase 2: flash attention =================
// grid (32, HEADS, B), block 128 (4 waves); each wave owns TWO 16-row Q tiles
// so every K/V operand fetched feeds two WMMAs.
__global__ void __launch_bounds__(128)
attn_kernel(const __bf16* __restrict__ Qs, const __bf16* __restrict__ Ks,
            const __bf16* __restrict__ Vt, __bf16* __restrict__ At) {
  __shared__ __bf16 Plds[4 * 2 * 16 * 32];  // per-wave two P tiles
  const int t    = threadIdx.x;
  const int wav  = t >> 5;
  const int lane = t & 31;
  const int col  = lane & 15;
  const int half = lane >> 4;
  const int h    = blockIdx.y;
  const int b    = blockIdx.z;
  const int mbase = (blockIdx.x * 4 + wav) * 32;  // 32 query rows per wave
  const size_t bh = (size_t)(b * HEADS + h);

  const __bf16* qb = Qs + bh * NPIX * DIMH;
  const __bf16* kb = Ks + bh * NPIX * DIMH;
  const __bf16* vb = Vt + bh * DIMH * NPIX;

  // Two Q tiles in A layout, register-resident for the whole KV sweep.
  v16bf Aq[2];
#pragma unroll
  for (int u = 0; u < 2; ++u) {
    const __bf16* qr = qb + (size_t)(mbase + 16 * u + col) * DIMH;
    Aq[u] = ld_bf16_2x8(qr + 16 * half, qr + 16 * half + 8);
  }

  v8f O0[2] = {}, O1[2] = {};
  float mi[2][8], li[2][8];
#pragma unroll
  for (int u = 0; u < 2; ++u)
#pragma unroll
    for (int g = 0; g < 8; ++g) { mi[u][g] = -1e30f; li[u][g] = 0.f; }

  __bf16* pw = Plds + wav * 1024;  // wave-private: two 16x32 tiles

  for (int kv = 0; kv < NPIX; kv += 32) {
    // ---- scores S = Q * K^T (contraction over d, innermost in Ks rows) ----
    v16bf Bk0 = ld_bf16x16(kb + (size_t)(kv + col) * DIMH + 16 * half);
    v16bf Bk1 = ld_bf16x16(kb + (size_t)(kv + 16 + col) * DIMH + 16 * half);
    v8f z = {};
    v8f S0[2], S1[2];
#pragma unroll
    for (int u = 0; u < 2; ++u) {
      S0[u] = wmma_bf16(Aq[u], Bk0, z);
      S1[u] = wmma_bf16(Aq[u], Bk1, z);
    }

    // ---- online softmax; each row lives in 16 lanes of one wave half ----
#pragma unroll
    for (int u = 0; u < 2; ++u) {
#pragma unroll
      for (int g = 0; g < 8; ++g) {
        float rmax = fmaxf(S0[u][g], S1[u][g]);
#pragma unroll
        for (int off = 8; off >= 1; off >>= 1)
          rmax = fmaxf(rmax, __shfl_xor(rmax, off, 16));
        const float mn    = fmaxf(mi[u][g], rmax);
        const float alpha = __expf(mi[u][g] - mn);
        const float e0    = __expf(S0[u][g] - mn);
        const float e1    = __expf(S1[u][g] - mn);
        float rsum = e0 + e1;
#pragma unroll
        for (int off = 8; off >= 1; off >>= 1)
          rsum += __shfl_xor(rsum, off, 16);
        li[u][g] = li[u][g] * alpha + rsum;
        mi[u][g] = mn;
        O0[u][g] *= alpha;
        O1[u][g] *= alpha;
        const int row = g + 8 * half;  // C-layout row owned by this lane
        pw[u * 512 + row * 32 + col]      = (__bf16)e0;
        pw[u * 512 + row * 32 + 16 + col] = (__bf16)e1;
      }
    }
    lds_compiler_fence();  // P stores precede cross-lane A-layout reads

    // ---- O += P * V (contraction over n, innermost in Vt rows) ----
    v16bf Bv0 = ld_bf16x16(vb + (size_t)col * NPIX + kv + 16 * half);
    v16bf Bv1 = ld_bf16x16(vb + (size_t)(16 + col) * NPIX + kv + 16 * half);
#pragma unroll
    for (int u = 0; u < 2; ++u) {
      const __bf16* pr = pw + u * 512 + col * 32;
      v16bf Ap = ld_bf16_2x8(pr + 16 * half, pr + 16 * half + 8);
      O0[u] = wmma_bf16(Ap, Bv0, O0[u]);
      O1[u] = wmma_bf16(Ap, Bv1, O1[u]);
    }
    lds_compiler_fence();  // A-reads precede next iteration's P stores
  }

  // ---- normalize, write transposed attention output At[b, p, h*32+d] ----
#pragma unroll
  for (int u = 0; u < 2; ++u) {
#pragma unroll
    for (int g = 0; g < 8; ++g) {
      const float inv = 1.f / li[u][g];
      const int p = mbase + 16 * u + g + 8 * half;
      __bf16* dst = At + ((size_t)b * NPIX + p) * INNER + h * DIMH;
      dst[col]      = (__bf16)(O0[u][g] * inv);
      dst[16 + col] = (__bf16)(O1[u][g] * inv);
    }
  }
}

// ================= Phase 3: output projection + bias =================
// grid (256 p-tiles, B), block 256 (8 waves); wave w handles co-tile w.
__global__ void __launch_bounds__(256)
proj_kernel(const __bf16* __restrict__ At, const float* __restrict__ wproj,
            const float* __restrict__ bias, float* __restrict__ out) {
  const int t    = threadIdx.x;
  const int wav  = t >> 5;
  const int lane = t & 31;
  const int col  = lane & 15;
  const int half = lane >> 4;
  const int b    = blockIdx.y;
  const int p0   = blockIdx.x * 16;
  const int cobase = wav * 16;

  v8f acc = {};
#pragma unroll
  for (int kc = 0; kc < INNER; kc += 32) {
    const float* wr = wproj + (size_t)(cobase + col) * INNER + kc;
    v16bf A = ld_f32_to_bf16(wr + 16 * half, wr + 16 * half + 8);
    v16bf B = ld_bf16x16(At + ((size_t)b * NPIX + p0 + col) * INNER + kc + 16 * half);
    acc = wmma_bf16(A, B, acc);
  }
#pragma unroll
  for (int g = 0; g < 8; ++g) {
    const int co = cobase + g + 8 * half;
    out[((size_t)b * INNER + co) * NPIX + p0 + col] = acc[g] + bias[co];
  }
}

// ================= host launcher =================
extern "C" void kernel_launch(void* const* d_in, const int* in_sizes, int n_in,
                              void* d_out, int out_size, void* d_ws, size_t ws_size,
                              hipStream_t stream) {
  const float* x     = (const float*)d_in[0];
  const float* wqkv  = (const float*)d_in[1];
  const float* wproj = (const float*)d_in[2];
  const float* bias  = (const float*)d_in[3];
  float* out = (float*)d_out;

  const size_t nelem = (size_t)BATCH * HEADS * NPIX * DIMH;  // 2M bf16 per buffer
  __bf16* Qs = (__bf16*)d_ws;
  __bf16* Ks = Qs + nelem;
  __bf16* Vt = Ks + nelem;
  __bf16* At = Vt + nelem;  // [B, NPIX, INNER], same element count

  qkv_kernel<<<dim3(NPIX / 64, 3, BATCH), 256, 0, stream>>>(x, wqkv, Qs, Ks, Vt);
  attn_kernel<<<dim3(NPIX / 128, HEADS, BATCH), 128, 0, stream>>>(Qs, Ks, Vt, At);
  proj_kernel<<<dim3(NPIX / 16, BATCH), 256, 0, stream>>>(At, wproj, bias, out);
}
